// MHA_26414048870518
// MI455X (gfx1250) — matrix operand order repaired and checked by reading, not compiled, not run on hardware
//
#include <hip/hip_runtime.h>
#include <hip/hip_bf16.h>

#define EMB   1024
#define HEADS 16
#define HDIM  64
#define BATCH 4
#define SEQ   2048
#define ATT_SCALE 0.125f   // 1/sqrt(64)

typedef __bf16 bf16;
typedef __attribute__((ext_vector_type(16))) __bf16 v16bf;
typedef __attribute__((ext_vector_type(8)))  float  v8f;
typedef __attribute__((ext_vector_type(4)))  int    v4i;

#define WMMA_BF16(A, B, C) \
  __builtin_amdgcn_wmma_f32_16x16x32_bf16(false, (A), false, (B), (short)0, (C), false, false)

// ---- gfx1250 async global->LDS path (guarded; falls back to reg-staged copy) ----
#if defined(__has_builtin)
#if __has_builtin(__builtin_amdgcn_global_load_async_to_lds_b128)
#define HAVE_ASYNC_LDS 1
#endif
#endif

static __device__ inline void async_wait_all() {
#if defined(HAVE_ASYNC_LDS)
#if __has_builtin(__builtin_amdgcn_s_wait_asynccnt)
  __builtin_amdgcn_s_wait_asynccnt(0);
#else
  asm volatile("s_wait_asynccnt 0" ::: "memory");
#endif
#endif
}

// Copy one 16-byte chunk global -> LDS.
static __device__ inline void copy16_g2l(const bf16* g, bf16* l) {
#if defined(HAVE_ASYNC_LDS)
  __builtin_amdgcn_global_load_async_to_lds_b128((v4i*)g, (v4i*)l, 0, 0);
#else
  *(uint4*)l = *(const uint4*)g;
#endif
}

// Stage a 64(row) x 32(k) bf16 weight tile (4KB) into LDS with 128 threads.
// gsrc points at row 0, k-chunk start; rows stride EMB elements.
static __device__ inline void stage_btile(const bf16* gsrc, bf16* lbuf, int t) {
#pragma unroll
  for (int c = 0; c < 2; ++c) {
    const int ch  = t * 2 + c;        // 0..255 chunks of 16B
    const int row = ch >> 2;
    const int seg = (ch & 3) * 8;     // 8 bf16 per 16B
    copy16_g2l(gsrc + (size_t)row * EMB + seg, lbuf + row * 32 + seg);
  }
}

// ---- fragment builders (16-bit WMMA layouts, wave32) ----
// A-fragment (16x32, row = lane&15): half 0 -> K {0..7,16..23}, half 1 -> K {8..15,24..31}
static __device__ inline v16bf fragA_bf16(const bf16* p, int half) {
  v16bf f;
  const bf16* p0 = p + 8 * half;
  const bf16* p1 = p + 16 + 8 * half;
#pragma unroll
  for (int i = 0; i < 8; ++i) { f[i] = p0[i]; f[8 + i] = p1[i]; }
  return f;
}
static __device__ inline v16bf fragA_f32(const float* p, int half) {
  v16bf f;
  const float* p0 = p + 8 * half;
  const float* p1 = p + 16 + 8 * half;
#pragma unroll
  for (int i = 0; i < 8; ++i) { f[i] = (bf16)p0[i]; f[8 + i] = (bf16)p1[i]; }
  return f;
}
// B-fragment (32x16, col = lane&15): half 0 -> K {0..7,16..23}, half 1 -> K {8..15,24..31}
static __device__ inline v16bf fragB_bf16(const bf16* p, int half) {
  v16bf f;
  const bf16* p0 = p + 8 * half;
  const bf16* p1 = p + 16 + 8 * half;
#pragma unroll
  for (int i = 0; i < 8; ++i) { f[i] = p0[i]; f[8 + i] = p1[i]; }
  return f;
}

// ---------------- kernel 0: pack weights fp32 [K,N] -> bf16 WT[p][N][K] ----------------
__global__ __launch_bounds__(256) void mha_pack_weights(
    const float* __restrict__ WQ, const float* __restrict__ WK,
    const float* __restrict__ WV, const float* __restrict__ WO,
    bf16* __restrict__ WT) {
  __shared__ float tile[32][33];
  const int p  = blockIdx.z;
  const float* W = p == 0 ? WQ : (p == 1 ? WK : (p == 2 ? WV : WO));
  const int k0 = blockIdx.x * 32;
  const int n0 = blockIdx.y * 32;
  const int tx = threadIdx.x & 31;
  const int ty = threadIdx.x >> 5;   // 0..7
#pragma unroll
  for (int r = 0; r < 32; r += 8)
    tile[ty + r][tx] = W[(size_t)(k0 + ty + r) * EMB + n0 + tx];
  __syncthreads();
  bf16* dst = WT + ((size_t)p * EMB + n0) * EMB + k0;
#pragma unroll
  for (int r = 0; r < 32; r += 8)
    dst[(size_t)(ty + r) * EMB + tx] = (bf16)tile[tx][ty + r];
}

// ---------------- kernel 1: fused Q/K/V projection ----------------
// grid (128, 16, 3), block 128 (4 waves). Block shares one 64-col B stripe,
// async-staged in LDS (double-buffered); each wave owns a 16x64 C-tile.
__global__ __launch_bounds__(128) void mha_proj_qkv(
    const float* __restrict__ Qi, const float* __restrict__ Ki, const float* __restrict__ Vi,
    const float* __restrict__ bQ, const float* __restrict__ bK, const float* __restrict__ bV,
    const bf16* __restrict__ WT,
    bf16* __restrict__ qb, bf16* __restrict__ kb, bf16* __restrict__ vT) {
  __shared__ __align__(16) bf16 Wt[2][64 * 32];
  const int t    = threadIdx.x;
  const int lane = t & 31;
  const int wid  = t >> 5;
  const int half = lane >> 4;
  const int lc   = lane & 15;
  const int m0   = (blockIdx.x * 4 + wid) * 16;
  const int n0   = blockIdx.y * 64;
  const int which = blockIdx.z;

  const float* X    = which == 0 ? Qi : (which == 1 ? Ki : Vi);
  const float* bias = which == 0 ? bQ : (which == 1 ? bK : bV);
  const bf16*  Wp   = WT + (size_t)which * EMB * EMB + (size_t)n0 * EMB;  // rows n0..n0+63

  const float* ap = X + (size_t)(m0 + lc) * EMB;

  stage_btile(Wp, Wt[0], t);
  async_wait_all();
  __syncthreads();

  v8f acc0 = {}, acc1 = {}, acc2 = {}, acc3 = {};
  int buf = 0;
  for (int k0 = 0; k0 < EMB; k0 += 32, buf ^= 1) {
    if (k0 + 32 < EMB) stage_btile(Wp + (k0 + 32), Wt[buf ^ 1], t);
    const v16bf a = fragA_f32(ap + k0, half);
    const bf16* lb = Wt[buf];
    acc0 = WMMA_BF16(a, fragB_bf16(lb + (0  + lc) * 32, half), acc0);
    acc1 = WMMA_BF16(a, fragB_bf16(lb + (16 + lc) * 32, half), acc1);
    acc2 = WMMA_BF16(a, fragB_bf16(lb + (32 + lc) * 32, half), acc2);
    acc3 = WMMA_BF16(a, fragB_bf16(lb + (48 + lc) * 32, half), acc3);
    async_wait_all();
    __syncthreads();
  }

#pragma unroll
  for (int j = 0; j < 4; ++j) {
    const v8f* acc = j == 0 ? &acc0 : (j == 1 ? &acc1 : (j == 2 ? &acc2 : &acc3));
    const int ocol = n0 + 16 * j + lc;
    const float bv = bias[ocol];
    const int h_ = ocol >> 6, d_ = ocol & (HDIM - 1);
#pragma unroll
    for (int r = 0; r < 8; ++r) {
      const int orow = m0 + r + 8 * half;
      const int b_ = orow >> 11, l_ = orow & (SEQ - 1);
      const size_t bh = (size_t)b_ * HEADS + h_;
      const bf16 o = (bf16)((*acc)[r] + bv);
      if (which == 2)      vT[(bh * HDIM + d_) * SEQ + l_] = o;   // v stored [B,H,D,L]
      else if (which == 1) kb[(bh * SEQ + l_) * HDIM + d_] = o;   // [B,H,L,D]
      else                 qb[(bh * SEQ + l_) * HDIM + d_] = o;   // [B,H,L,D]
    }
  }
}

// ---------------- kernel 2: causal flash attention ----------------
// grid (32, 64), block 128: wave handles one 16-row q tile of one (b,h)
__global__ __launch_bounds__(128) void mha_flash_attn(
    const bf16* __restrict__ qb, const bf16* __restrict__ kb,
    const bf16* __restrict__ vT, bf16* __restrict__ attnb) {
  __shared__ bf16 Pl[4][16 * 32];
  const int lane = threadIdx.x & 31;
  const int wid  = threadIdx.x >> 5;
  const int half = lane >> 4;
  const int lc   = lane & 15;
  const int bh   = blockIdx.y;
  const int q0   = (blockIdx.x * 4 + wid) * 16;
  bf16* P = Pl[wid];

  const bf16* qp = qb + ((size_t)bh * SEQ + q0 + lc) * HDIM;
  const v16bf Aq0 = fragA_bf16(qp, half);        // d 0..31
  const v16bf Aq1 = fragA_bf16(qp + 32, half);   // d 32..63

  v8f O0 = {}, O1 = {}, O2 = {}, O3 = {};
  float mrun[8], lrun[8];
#pragma unroll
  for (int r = 0; r < 8; ++r) { mrun[r] = -1e30f; lrun[r] = 0.0f; }

  const bf16* kbase = kb + (size_t)bh * SEQ * HDIM;
  const bf16* vbase = vT + ((size_t)bh * HDIM + lc) * SEQ;

  for (int k0 = 0; k0 < q0 + 16; k0 += 32) {
    if (k0 + 32 < q0 + 16) {   // prefetch next block into cache
      __builtin_prefetch(kbase + (size_t)(k0 + 32 + lc) * HDIM, 0, 1);
      __builtin_prefetch(vbase + k0 + 32, 0, 1);
    }
    // ---- S = q . k^T  (two 16x16 tiles, K=64 in two 32-chunks) ----
    const bf16* kp0 = kbase + (size_t)(k0 + lc) * HDIM;
    const bf16* kp1 = kbase + (size_t)(k0 + 16 + lc) * HDIM;
    v16bf B00 = fragB_bf16(kp0, half);
    v16bf B01 = fragB_bf16(kp0 + 32, half);
    v16bf B10 = fragB_bf16(kp1, half);
    v16bf B11 = fragB_bf16(kp1 + 32, half);
    v8f S0 = {}, S1 = {};
    S0 = WMMA_BF16(Aq0, B00, S0);
    S0 = WMMA_BF16(Aq1, B01, S0);
    S1 = WMMA_BF16(Aq0, B10, S1);
    S1 = WMMA_BF16(Aq1, B11, S1);

    // ---- online softmax (fp32; row reductions across 16-lane C groups) ----
    float p0[8], p1[8], alpha[8];
#pragma unroll
    for (int r = 0; r < 8; ++r) {
      const int qrow = q0 + r + 8 * half;
      const int c0 = k0 + lc, c1 = c0 + 16;
      float s0 = (c0 <= qrow) ? S0[r] * ATT_SCALE : -1e30f;
      float s1 = (c1 <= qrow) ? S1[r] * ATT_SCALE : -1e30f;
      float mx = fmaxf(s0, s1);
      mx = fmaxf(mx, __shfl_xor(mx, 1, 32));
      mx = fmaxf(mx, __shfl_xor(mx, 2, 32));
      mx = fmaxf(mx, __shfl_xor(mx, 4, 32));
      mx = fmaxf(mx, __shfl_xor(mx, 8, 32));
      const float mnew = fmaxf(mrun[r], mx);
      alpha[r] = __expf(mrun[r] - mnew);
      p0[r] = __expf(s0 - mnew);
      p1[r] = __expf(s1 - mnew);
      float rs = p0[r] + p1[r];
      rs += __shfl_xor(rs, 1, 32);
      rs += __shfl_xor(rs, 2, 32);
      rs += __shfl_xor(rs, 4, 32);
      rs += __shfl_xor(rs, 8, 32);
      lrun[r] = lrun[r] * alpha[r] + rs;
      mrun[r] = mnew;
    }
#pragma unroll
    for (int r = 0; r < 8; ++r) {
      O0[r] *= alpha[r]; O1[r] *= alpha[r]; O2[r] *= alpha[r]; O3[r] *= alpha[r];
    }

    // ---- repack P (C-layout -> A-layout) through LDS ----
#pragma unroll
    for (int r = 0; r < 8; ++r) {
      const int prow = r + 8 * half;
      P[prow * 32 + lc]      = (bf16)p0[r];
      P[prow * 32 + 16 + lc] = (bf16)p1[r];
    }
    asm volatile("s_wait_dscnt 0" ::: "memory");
    const v16bf Pf = fragA_bf16(P + lc * 32, half);

    // ---- O += P . V  (V stored [B,H,D,L]: contiguous keys per lane) ----
    const bf16* vp = vbase + k0;
    {
      v16bf Vf;
      Vf = fragB_bf16(vp + (size_t)0 * 16 * SEQ, half); O0 = WMMA_BF16(Pf, Vf, O0);
      Vf = fragB_bf16(vp + (size_t)1 * 16 * SEQ, half); O1 = WMMA_BF16(Pf, Vf, O1);
      Vf = fragB_bf16(vp + (size_t)2 * 16 * SEQ, half); O2 = WMMA_BF16(Pf, Vf, O2);
      Vf = fragB_bf16(vp + (size_t)3 * 16 * SEQ, half); O3 = WMMA_BF16(Pf, Vf, O3);
    }
  }

  // ---- epilogue: normalize and write concat-head layout [B,L,E] ----
  const int b_ = bh >> 4, h_ = bh & 15;
#pragma unroll
  for (int r = 0; r < 8; ++r) {
    const int row = q0 + r + 8 * half;
    const float inv = 1.0f / lrun[r];
    const size_t base = ((size_t)b_ * SEQ + row) * EMB + h_ * HDIM + lc;
    attnb[base + 0]  = (bf16)(O0[r] * inv);
    attnb[base + 16] = (bf16)(O1[r] * inv);
    attnb[base + 32] = (bf16)(O2[r] * inv);
    attnb[base + 48] = (bf16)(O3[r] * inv);
  }
}

// ---------------- kernel 3: output projection ----------------
// grid (128, 16), block 128 (4 waves); LDS-staged B like kernel 1, fp32 out.
__global__ __launch_bounds__(128) void mha_proj_out(
    const bf16* __restrict__ attnb, const bf16* __restrict__ WT,
    const float* __restrict__ bO, float* __restrict__ out) {
  __shared__ __align__(16) bf16 Wt[2][64 * 32];
  const int t    = threadIdx.x;
  const int lane = t & 31;
  const int wid  = t >> 5;
  const int half = lane >> 4;
  const int lc   = lane & 15;
  const int m0   = (blockIdx.x * 4 + wid) * 16;
  const int n0   = blockIdx.y * 64;

  const bf16* Wp = WT + (size_t)3 * EMB * EMB + (size_t)n0 * EMB;  // WO^T rows n0..n0+63
  const bf16* ap = attnb + (size_t)(m0 + lc) * EMB;

  stage_btile(Wp, Wt[0], t);
  async_wait_all();
  __syncthreads();

  v8f acc0 = {}, acc1 = {}, acc2 = {}, acc3 = {};
  int buf = 0;
  for (int k0 = 0; k0 < EMB; k0 += 32, buf ^= 1) {
    if (k0 + 32 < EMB) stage_btile(Wp + (k0 + 32), Wt[buf ^ 1], t);
    const v16bf a = fragA_bf16(ap + k0, half);
    const bf16* lb = Wt[buf];
    acc0 = WMMA_BF16(a, fragB_bf16(lb + (0  + lc) * 32, half), acc0);
    acc1 = WMMA_BF16(a, fragB_bf16(lb + (16 + lc) * 32, half), acc1);
    acc2 = WMMA_BF16(a, fragB_bf16(lb + (32 + lc) * 32, half), acc2);
    acc3 = WMMA_BF16(a, fragB_bf16(lb + (48 + lc) * 32, half), acc3);
    async_wait_all();
    __syncthreads();
  }

#pragma unroll
  for (int j = 0; j < 4; ++j) {
    const v8f* acc = j == 0 ? &acc0 : (j == 1 ? &acc1 : (j == 2 ? &acc2 : &acc3));
    const int ocol = n0 + 16 * j + lc;
    const float bv = bO[ocol];
#pragma unroll
    for (int r = 0; r < 8; ++r)
      out[(size_t)(m0 + r + 8 * half) * EMB + ocol] = (*acc)[r] + bv;
  }
}

extern "C" void kernel_launch(void* const* d_in, const int* in_sizes, int n_in,
                              void* d_out, int out_size, void* d_ws, size_t ws_size,
                              hipStream_t stream) {
  (void)in_sizes; (void)n_in; (void)out_size; (void)ws_size;
  const float* Q  = (const float*)d_in[0];
  const float* K  = (const float*)d_in[1];
  const float* V  = (const float*)d_in[2];
  const float* WQ = (const float*)d_in[3];
  const float* bQ = (const float*)d_in[4];
  const float* WK = (const float*)d_in[5];
  const float* bK = (const float*)d_in[6];
  const float* WV = (const float*)d_in[7];
  const float* bV = (const float*)d_in[8];
  const float* WO = (const float*)d_in[9];
  const float* bO = (const float*)d_in[10];

  const size_t nTok = (size_t)BATCH * SEQ * EMB;  // 8M elems
  bf16* qb    = (bf16*)d_ws;
  bf16* kb    = qb + nTok;
  bf16* vT    = kb + nTok;
  bf16* attnb = vT + nTok;
  bf16* WT    = attnb + nTok;                     // 4 * 1024 * 1024 bf16

  mha_pack_weights<<<dim3(32, 32, 4), 256, 0, stream>>>(WQ, WK, WV, WO, WT);
  mha_proj_qkv<<<dim3(128, 16, 3), 128, 0, stream>>>(
      Q, K, V, bQ, bK, bV, WT, qb, kb, vT);
  mha_flash_attn<<<dim3(32, 64), 128, 0, stream>>>(qb, kb, vT, attnb);
  mha_proj_out<<<dim3(128, 16), 128, 0, stream>>>(attnb, WT, bO, (float*)d_out);
}
